// DCN_73409581023645
// MI455X (gfx1250) — matrix operand order repaired and checked by reading, not compiled, not run on hardware
//
#include <hip/hip_runtime.h>
#include <hip/hip_bf16.h>

typedef __attribute__((ext_vector_type(16))) _Float16 v16h;
typedef __attribute__((ext_vector_type(8)))  _Float16 v8h;
typedef __attribute__((ext_vector_type(8)))  float    v8f;

#define B_    4
#define C_    64
#define H_    128
#define W_    128
#define CO_   64
#define KTAPS 9
#define HW_   (H_*W_)              // 16384
#define NOFF  18                   // 2*KTAPS

// ---- workspace layout (bytes) ----
#define XT_BYTES   ((size_t)B_*HW_*C_*4)          // 16 MB  NHWC f32
#define OFF_BYTES  ((size_t)B_*HW_*KTAPS*2*4)     // 4.5 MB [b][pos][tap][dy,dx]
#define WTF_BYTES  ((size_t)KTAPS*2*4*32*16*2)    // f16 fragment-native B tiles
#define OWT_BYTES  ((size_t)KTAPS*C_*NOFF*4)      // f32 [tap][c][18]
#define XT_OFF   ((size_t)0)
#define OFF_OFF  (XT_OFF + XT_BYTES)
#define WTF_OFF  (OFF_OFF + OFF_BYTES)
#define OWT_OFF  ((WTF_OFF + WTF_BYTES + 255) & ~(size_t)255)

// ---------------- kernel 1: NCHW -> NHWC transpose of x ----------------
__global__ void k_transpose_x(const float* __restrict__ x, float* __restrict__ xT) {
    int idx = blockIdx.x * blockDim.x + threadIdx.x;   // [b][y][x][c], c fastest
    int c   = idx & 63;
    int pix = idx >> 6;            // b*HW + y*W + x
    int b   = pix >> 14;
    int yx  = pix & (HW_ - 1);
    xT[idx] = x[((size_t)(b * C_ + c)) * HW_ + yx];
}

// ---------------- kernel 2: weight prep ----------------
// wTf: fragment-native B layout.  Fragment element e of lane l for chunk
// (tap, half, nsub) is  K=laneHi*16+e (within 32-chunk), N=nsub*16+(l&15):
//   wTf[(((tap*2+half)*4+nsub)*32 + l)*16 + e] = w[oc][c][tap]
__global__ void k_prep_w(const float* __restrict__ w, const float* __restrict__ ow,
                         _Float16* __restrict__ wTf, float* __restrict__ owT) {
    int idx = blockIdx.x * blockDim.x + threadIdx.x;
    if (idx < KTAPS * 2 * 4 * 32 * 16) {
        int e    = idx & 15;
        int lane = (idx >> 4) & 31;
        int nsub = (idx >> 9) & 3;
        int half = (idx >> 11) & 1;
        int tap  = idx >> 12;
        int oc   = nsub * 16 + (lane & 15);
        int c    = half * 32 + ((lane >> 4) ? 16 : 0) + e;
        wTf[idx] = (_Float16)w[((size_t)(oc * C_ + c)) * KTAPS + tap];
    }
    if (idx < KTAPS * C_ * NOFF) {                // owT[(tap*64+c)*18+oc] = ow[oc][c][tap]
        int oc = idx % NOFF; int t = idx / NOFF;
        int c = t & 63; int tap = t >> 6;
        owT[idx] = ow[((size_t)(oc * C_ + c)) * KTAPS + tap];
    }
}

// ---------------- kernel 3: offset conv (18 channels) ----------------
__global__ void k_offset_conv(const float* __restrict__ xT, const float* __restrict__ owT,
                              const float* __restrict__ obias, float* __restrict__ offBuf) {
    __shared__ float sw[KTAPS * C_ * NOFF];       // 41472 B of 320 KB
    for (int i = threadIdx.x; i < KTAPS * C_ * NOFF; i += blockDim.x) sw[i] = owT[i];
    __syncthreads();

    int gid = blockIdx.x * blockDim.x + threadIdx.x;   // b*HW + pos
    int b  = gid >> 14;
    int yx = gid & (HW_ - 1);
    int ho = yx >> 7, wo = yx & (W_ - 1);

    float acc[NOFF];
#pragma unroll
    for (int i = 0; i < NOFF; i++) acc[i] = obias[i];

    for (int ky = 0; ky < 3; ky++) {
        int y = ho - 1 + ky;
        if ((unsigned)y >= (unsigned)H_) continue;
        for (int kx = 0; kx < 3; kx++) {
            int x = wo - 1 + kx;
            if ((unsigned)x >= (unsigned)W_) continue;
            int tap = ky * 3 + kx;
            const float* xp = xT + ((size_t)(b * HW_ + y * W_ + x)) * C_;
            const float* wp = sw + tap * C_ * NOFF;
#pragma unroll 4
            for (int c = 0; c < C_; c++) {
                float xv = xp[c];
#pragma unroll
                for (int oc = 0; oc < NOFF; oc++) acc[oc] += xv * wp[c * NOFF + oc];
            }
        }
    }
    // channel oc = 2*tap + j  (j=0 -> dy, j=1 -> dx); matches [tap][dy,dx] layout
    float* op = offBuf + (size_t)gid * NOFF;
#pragma unroll
    for (int i = 0; i < NOFF; i++) op[i] = acc[i];
}

// ---------------- kernel 4: deformable sampling + WMMA GEMM ----------------
#define MTILE 128
#define LDSS  72     // patch row stride (halves): 144 B, 16B-aligned, conflict-free b128
#define OTS   129    // out-staging row stride (floats): bank (oc+m)%64

__global__ void __launch_bounds__(256)
k_deform_wmma(const float* __restrict__ xT, const float* __restrict__ offBuf,
              const _Float16* __restrict__ wTf, const float* __restrict__ bias,
              float* __restrict__ out) {
    // union: patch tile (18432 B of f16) during taps, f32 out-staging (33024 B) after
    __shared__ __align__(16) unsigned char smem[CO_ * OTS * 4];
    _Float16* patch = (_Float16*)smem;
    float*    ot    = (float*)smem;

    int b       = blockIdx.x >> 7;                 // 128 tiles per batch
    int posBase = (blockIdx.x & 127) * MTILE;

    int tid    = threadIdx.x;
    int lane   = tid & 31;
    int wave   = tid >> 5;                         // 8 wave32
    int laneHi = (lane >> 4) & 1;
    int laneM  = lane & 15;

    int nsub   = wave & 3;                         // oc subtile
    int mgroup = wave >> 2;                        // rows [mgroup*64, +64)
    int ocCol  = nsub * 16 + laneM;

    float bv = bias[ocCol];
    v8f acc[4];
#pragma unroll
    for (int s = 0; s < 4; s++)
#pragma unroll
        for (int r = 0; r < 8; r++) acc[s][r] = bv;

    // LDS-fill role: one position, 32 channels per thread
    int fm    = tid & 127;
    int fhalf = tid >> 7;
    int gpos  = posBase + fm;
    int ho = gpos >> 7, wo = gpos & (W_ - 1);

    for (int tap = 0; tap < KTAPS; tap++) {
        int ky = tap / 3, kx = tap % 3;

        __syncthreads();                           // protect prior-iteration reads
        {
            const float* offp = offBuf + (((size_t)(b * HW_ + gpos)) * KTAPS + tap) * 2;
            float dy = offp[0], dx = offp[1];
            float py = (float)(ho - 1 + ky) + dy;
            float px = (float)(wo - 1 + kx) + dx;
            float fy = floorf(py), fx = floorf(px);
            float ly = py - fy, lx = px - fx;
            int y0 = (int)fy, x0 = (int)fx;
            int y1 = y0 + 1, x1 = x0 + 1;
            bool vy0 = (unsigned)y0 < (unsigned)H_, vy1 = (unsigned)y1 < (unsigned)H_;
            bool vx0 = (unsigned)x0 < (unsigned)W_, vx1 = (unsigned)x1 < (unsigned)W_;
            int y0c = min(max(y0, 0), H_ - 1), y1c = min(max(y1, 0), H_ - 1);
            int x0c = min(max(x0, 0), W_ - 1), x1c = min(max(x1, 0), W_ - 1);
            float w00 = (vy0 && vx0) ? (1.f - ly) * (1.f - lx) : 0.f;
            float w01 = (vy0 && vx1) ? (1.f - ly) * lx         : 0.f;
            float w10 = (vy1 && vx0) ? ly * (1.f - lx)         : 0.f;
            float w11 = (vy1 && vx1) ? ly * lx                 : 0.f;
            size_t bb = (size_t)b * HW_;
            const float* p00 = xT + (bb + (size_t)y0c * W_ + x0c) * C_ + fhalf * 32;
            const float* p01 = xT + (bb + (size_t)y0c * W_ + x1c) * C_ + fhalf * 32;
            const float* p10 = xT + (bb + (size_t)y1c * W_ + x0c) * C_ + fhalf * 32;
            const float* p11 = xT + (bb + (size_t)y1c * W_ + x1c) * C_ + fhalf * 32;
            _Float16* dst = patch + fm * LDSS + fhalf * 32;
#pragma unroll
            for (int cc = 0; cc < 32; cc++) {
                float v = p00[cc] * w00 + p01[cc] * w01 + p10[cc] * w10 + p11[cc] * w11;
                // chunk stored as [K0-31] contiguous -> lanes read contiguously
                int p = cc;
                dst[p] = (_Float16)v;
            }
        }
        __syncthreads();

#pragma unroll
        for (int half = 0; half < 2; half++) {
            // B fragment: one contiguous 32-byte coalesced load per lane
            v16h bfrag = *(const v16h*)(wTf + ((size_t)(((tap * 2 + half) * 4 + nsub) * 32 + lane)) * 16);
#pragma unroll
            for (int s = 0; s < 4; s++) {
                const _Float16* ap = patch + (mgroup * 64 + s * 16 + laneM) * LDSS
                                   + half * 32 + laneHi * 16;
                union { v16h v; v8h h[2]; } au;
                au.h[0] = *(const v8h*)(ap);
                au.h[1] = *(const v8h*)(ap + 8);
                acc[s] = __builtin_amdgcn_wmma_f32_16x16x32_f16(
                    false, au.v, false, bfrag, (short)0, acc[s], false, false);
            }
        }
    }

    // ---- stage D tiles in LDS, then coalesced writeout ----
    __syncthreads();                               // done reading patch
#pragma unroll
    for (int s = 0; s < 4; s++)
#pragma unroll
        for (int r = 0; r < 8; r++) {
            int m = mgroup * 64 + s * 16 + laneHi * 8 + r;   // D: VGPR r -> M=r+8*laneHi
            ot[ocCol * OTS + m] = acc[s][r];
        }
    __syncthreads();
#pragma unroll
    for (int i = 0; i < (CO_ * MTILE) / 256; i++) {          // 32 coalesced iterations
        int fi  = i * 256 + tid;
        int oc  = fi >> 7;
        int pos = fi & 127;
        out[((size_t)(b * CO_ + oc)) * HW_ + (posBase + pos)] = ot[oc * OTS + pos];
    }
}

// ---------------- launcher ----------------
extern "C" void kernel_launch(void* const* d_in, const int* in_sizes, int n_in,
                              void* d_out, int out_size, void* d_ws, size_t ws_size,
                              hipStream_t stream) {
    const float* x     = (const float*)d_in[0];
    const float* wgt   = (const float*)d_in[1];
    const float* bias  = (const float*)d_in[2];
    const float* owgt  = (const float*)d_in[3];
    const float* obias = (const float*)d_in[4];
    float* out = (float*)d_out;

    char* ws = (char*)d_ws;
    float*    xT     = (float*)(ws + XT_OFF);
    float*    offBuf = (float*)(ws + OFF_OFF);
    _Float16* wTf    = (_Float16*)(ws + WTF_OFF);
    float*    owT    = (float*)(ws + OWT_OFF);

    // 1) x NCHW -> NHWC
    k_transpose_x<<<(B_ * HW_ * C_) / 256, 256, 0, stream>>>(x, xT);
    // 2) weight prep (fragment-native B layout + offset-conv weights)
    k_prep_w<<<(KTAPS * 2 * 4 * 32 * 16 + 255) / 256, 256, 0, stream>>>(wgt, owgt, wTf, owT);
    // 3) offset conv
    k_offset_conv<<<(B_ * HW_) / 256, 256, 0, stream>>>(xT, owT, obias, offBuf);
    // 4) deformable sampling + WMMA GEMM
    k_deform_wmma<<<B_ * (HW_ / MTILE), 256, 0, stream>>>(xT, offBuf, wTf, bias, out);
}
